// GRULayer_16088947491364
// MI455X (gfx1250) — matrix operand order repaired and checked by reading, not compiled, not run on hardware
//
#include <hip/hip_runtime.h>

// ---------------------------------------------------------------------------
// GRU (Keras reset_after=True), B=256, T=512, D=U=256, fused WMMA-bf16 scan.
// d_in: [0]=codes f32 [B,T,D], [1]=W f32 [D,3U], [2]=Uk f32 [U,3U], [3]=b f32 [2,3U]
// d_out: h_final f32 [B,U]
// d_ws: 768KB = W,Uk repacked to bf16 WMMA-B fragment layout.
//
// Design (recurrence-bound problem):
//  - each 16-row batch tile's scan is independent -> 16 blocks, zero grid sync
//  - wave w of 16 owns h columns [16w,16w+16) = one (z, r, h) gate tile triple
//  - state h kept fp32 in LDS; bf16 copies regenerated per step (RNE)
//  - weights pre-packed to bf16 WMMA-B fragment layout, L2-resident (768 KB)
//  - gates via v_exp_f32 / v_rcp_f32 (TRANS ops, co-execute with matrix pipe)
//  - kc loop unrolled 2x so next chunk's 12 global_load_b128 overlap current WMMAs
// ---------------------------------------------------------------------------

#define B_DIM 256
#define T_DIM 512
#define D_DIM 256
#define U_DIM 256
#define N3    768            // 3*U
#define NT3   48             // N3/16 gate column tiles
#define KC    8              // D/32 K-chunks
#define NFRAG (KC*NT3)       // 384 fragments per matrix
#define THREADS 512          // 16 waves; wave w owns h columns [16w,16w+16)

typedef __attribute__((ext_vector_type(16))) __bf16 v16bf;
typedef __attribute__((ext_vector_type(8)))  float  v8f;

union Frag {
    v16bf v;
    uint4 q[2];
};

// fp32 -> bf16 round-to-nearest-even, two packed into one dword
static __device__ __forceinline__ unsigned int pk2bf(float a, float b) {
    unsigned int ua = __float_as_uint(a), ub = __float_as_uint(b);
    unsigned int ra = (ua + 0x7FFFu + ((ua >> 16) & 1u)) >> 16;
    unsigned int rb = (ub + 0x7FFFu + ((ub >> 16) & 1u)) >> 16;
    return (ra & 0xFFFFu) | (rb << 16);
}

// fast sigmoid: v_mul + v_exp_f32 + v_add + v_rcp_f32 (TRANS pipe)
static __device__ __forceinline__ float fast_sigmoid(float x) {
    return __builtin_amdgcn_rcpf(1.0f + __expf(-x));
}

// ---------------------------------------------------------------------------
// Pre-pass: repack a [256, 768] fp32 matrix into bf16 WMMA B-fragments.
// Fragment (kc, nt) = 32x16 tile of B (K rows kc*32.., N cols nt*16..).
// B layout used here (16-bit, 32x16, contiguous K order matching A): lane L: N = L%16, K-half g = L/16;
// dword i (0..7) holds K = kc*32 + 16g + 2i (lo half) and +1 (hi half).
// Each lane's 8 dwords stored contiguously -> scan loads 2x b128 per frag.
// ---------------------------------------------------------------------------
__global__ void pack_b_kernel(const float* __restrict__ src,
                              unsigned int* __restrict__ dst) {
    const int frag = blockIdx.x;            // 0..383
    const int kc   = frag / NT3;
    const int nt   = frag % NT3;
    const int lane = threadIdx.x;           // 0..31
    const int n    = nt * 16 + (lane & 15);
    const int g    = lane >> 4;
#pragma unroll
    for (int i = 0; i < 8; ++i) {
        const int k = kc * 32 + 16 * g + 2 * i;
        dst[(size_t)frag * 256 + lane * 8 + i] =
            pk2bf(src[(size_t)k * N3 + n], src[(size_t)(k + 1) * N3 + n]);
    }
}

// ---------------------------------------------------------------------------
// Fused GRU scan. grid = 16 blocks (one per 16-row batch tile), 512 threads.
// ---------------------------------------------------------------------------
__launch_bounds__(THREADS, 1)
__global__ void gru_scan_kernel(const float* __restrict__ codes,
                                const float* __restrict__ bias,      // [2, 768]
                                const unsigned int* __restrict__ Wp, // packed W
                                const unsigned int* __restrict__ Up, // packed Uk
                                float* __restrict__ out) {
    __shared__ __align__(16) float          h_f32[16 * U_DIM]; // fp32 master state
    __shared__ __align__(16) unsigned short x_bf [16 * D_DIM]; // bf16 codes_t tile
    __shared__ __align__(16) unsigned short h_bf [16 * U_DIM]; // bf16 state copy
    __shared__ unsigned int rowmask;

    const int tid  = threadIdx.x;
    const int lane = tid & 31;
    const int w    = tid >> 5;              // wave 0..15 -> h cols [16w,16w+16)
    const int b0   = blockIdx.x * 16;       // batch tile base row

    // zero initial state
    for (int i = tid; i < 16 * U_DIM; i += THREADS) h_f32[i] = 0.0f;

    // per-lane gate biases (constant over time; same column for all 8 acc rows)
    const int nloc = lane & 15;
    const int g    = lane >> 4;
    const int ccol = w * 16 + nloc;         // h column this lane produces
    const float zb  = bias[ccol]             + bias[N3 + ccol];
    const float rb  = bias[256 + ccol]       + bias[N3 + 256 + ccol];
    const float hxb = bias[512 + ccol];
    const float hhb = bias[N3 + 512 + ccol];

    // cooperative load mapping: 512 threads cover 16 rows x 256 cols (8 each)
    const int lrow = tid >> 5;              // 0..15
    const int lcb  = (tid & 31) * 8;        // column base, 8 floats

    // per-wave B-fragment byte pointers; z tile at +0, r at +16KB, h at +32KB,
    // advance 48KB per K-chunk. All six loads use immediate offsets.
    const char* const wbase = (const char*)Wp + (size_t)w * 1024 + (size_t)lane * 32;
    const char* const ubase = (const char*)Up + (size_t)w * 1024 + (size_t)lane * 32;

#pragma unroll 1
    for (int t = 0; t < T_DIM; ++t) {
        __syncthreads();                    // prev-step h_f32 writes visible
        if (tid == 0) rowmask = 0;
        __syncthreads();

        // ---- stage codes[b0+lrow, t, lcb..lcb+8) -> bf16 LDS + mask bit ----
        {
            const float4* p = (const float4*)(codes +
                ((size_t)(b0 + lrow) * T_DIM + t) * D_DIM + lcb);
            unsigned int* xd = (unsigned int*)&x_bf[lrow * D_DIM + lcb];
            bool nz = false;
#pragma unroll
            for (int qi = 0; qi < 2; ++qi) {
                float4 v = p[qi];
                nz |= (v.x != 0.0f) | (v.y != 0.0f) | (v.z != 0.0f) | (v.w != 0.0f);
                xd[qi * 2 + 0] = pk2bf(v.x, v.y);
                xd[qi * 2 + 1] = pk2bf(v.z, v.w);
            }
            if (nz) atomicOr(&rowmask, 1u << lrow);
            if (t + 1 < T_DIM)              // global_prefetch_b8 next timestep
                __builtin_prefetch(codes +
                    ((size_t)(b0 + lrow) * T_DIM + (t + 1)) * D_DIM + lcb, 0, 1);
            // refresh bf16 copy of state
            const float* hr = &h_f32[lrow * U_DIM + lcb];
            unsigned int* hd = (unsigned int*)&h_bf[lrow * U_DIM + lcb];
#pragma unroll
            for (int qi = 0; qi < 4; ++qi)
                hd[qi] = pk2bf(hr[2 * qi], hr[2 * qi + 1]);
        }
        __syncthreads();
        const unsigned int mask = rowmask;

        // ---- accumulators seeded with biases ----
        v8f aZ, aR, aHX, aHH;
#pragma unroll
        for (int i = 0; i < 8; ++i) { aZ[i] = zb; aR[i] = rb; aHX[i] = hxb; aHH[i] = hhb; }

        const char* wp = wbase;
        const char* up = ubase;

        // ---- 16x256x48 slice: z, r, h tiles for this wave ----
        // unroll 2: scheduler hoists next chunk's independent loads over WMMAs
#pragma unroll 2
        for (int kc = 0; kc < KC; ++kc) {
            // A fragments (16x32 bf16): lane m=L%16 rows, K = kc*32 + 16g .. 16g+15
            Frag ax, ah;
            const int m = nloc;
            const uint4* xp = (const uint4*)&x_bf[m * D_DIM + kc * 32 + 16 * g];
            const uint4* hq = (const uint4*)&h_bf[m * U_DIM + kc * 32 + 16 * g];
            ax.q[0] = xp[0]; ax.q[1] = xp[1];   // contiguous 16 elements
            ah.q[0] = hq[0]; ah.q[1] = hq[1];

            Frag bwz, bwr, bwh, buz, bur, buh;
            bwz.q[0] = *(const uint4*)(wp +     0); bwz.q[1] = *(const uint4*)(wp +    16);
            bwr.q[0] = *(const uint4*)(wp + 16384); bwr.q[1] = *(const uint4*)(wp + 16400);
            bwh.q[0] = *(const uint4*)(wp + 32768); bwh.q[1] = *(const uint4*)(wp + 32784);
            buz.q[0] = *(const uint4*)(up +     0); buz.q[1] = *(const uint4*)(up +    16);
            bur.q[0] = *(const uint4*)(up + 16384); bur.q[1] = *(const uint4*)(up + 16400);
            buh.q[0] = *(const uint4*)(up + 32768); buh.q[1] = *(const uint4*)(up + 32784);

            // interleave so same-accumulator chains are spaced by independents
            aZ  = __builtin_amdgcn_wmma_f32_16x16x32_bf16(false, ax.v, false, bwz.v, (short)0, aZ,  false, false);
            aR  = __builtin_amdgcn_wmma_f32_16x16x32_bf16(false, ax.v, false, bwr.v, (short)0, aR,  false, false);
            aHX = __builtin_amdgcn_wmma_f32_16x16x32_bf16(false, ax.v, false, bwh.v, (short)0, aHX, false, false);
            aZ  = __builtin_amdgcn_wmma_f32_16x16x32_bf16(false, ah.v, false, buz.v, (short)0, aZ,  false, false);
            aR  = __builtin_amdgcn_wmma_f32_16x16x32_bf16(false, ah.v, false, bur.v, (short)0, aR,  false, false);
            aHH = __builtin_amdgcn_wmma_f32_16x16x32_bf16(false, ah.v, false, buh.v, (short)0, aHH, false, false);

            wp += NT3 * 1024;               // next K-chunk (48 KB stride)
            up += NT3 * 1024;
        }

        // ---- gates + state update (C/D layout: elem i -> row i+8g, col ccol) ----
#pragma unroll
        for (int i = 0; i < 8; ++i) {
            const int M = i + 8 * g;
            const float z  = fast_sigmoid(aZ[i]);
            const float rr = fast_sigmoid(aR[i]);
            const float e  = aHX[i] + rr * aHH[i];
            const float hc = 2.0f * fast_sigmoid(2.0f * e) - 1.0f;  // tanh(e)
            const float hp = h_f32[M * U_DIM + ccol];
            float hn = hc + z * (hp - hc);      // z*hp + (1-z)*hc
            hn = ((mask >> M) & 1u) ? hn : hp;  // masked step carries state
            h_f32[M * U_DIM + ccol] = hn;
        }
    }

    __syncthreads();
    // write h_final tile
#pragma unroll
    for (int qi = 0; qi < 8; ++qi)
        out[(size_t)(b0 + lrow) * U_DIM + lcb + qi] = h_f32[lrow * U_DIM + lcb + qi];
}

extern "C" void kernel_launch(void* const* d_in, const int* in_sizes, int n_in,
                              void* d_out, int out_size, void* d_ws, size_t ws_size,
                              hipStream_t stream) {
    const float* codes = (const float*)d_in[0];
    const float* W     = (const float*)d_in[1];
    const float* Uk    = (const float*)d_in[2];
    const float* bias  = (const float*)d_in[3];

    unsigned int* Wp = (unsigned int*)d_ws;              // 384 KB
    unsigned int* Up = Wp + (size_t)NFRAG * 256;         // 384 KB

    pack_b_kernel<<<NFRAG, 32, 0, stream>>>(W,  Wp);
    pack_b_kernel<<<NFRAG, 32, 0, stream>>>(Uk, Up);
    gru_scan_kernel<<<B_DIM / 16, THREADS, 0, stream>>>(codes, bias, Wp, Up,
                                                        (float*)d_out);
}
